// LinearAttention_82094004896062
// MI455X (gfx1250) — compile-verified, matrix operand order checked by reading, not run on hardware
//
#include <hip/hip_runtime.h>
#include <hip/hip_bf16.h>

#define DEV __device__ __forceinline__

typedef __attribute__((ext_vector_type(16))) _Float16 v16h;
typedef __attribute__((ext_vector_type(8)))  _Float16 v8h;
typedef __attribute__((ext_vector_type(8)))  float    v8f;

// Problem constants
constexpr int  BB   = 4, TT = 4096, DD = 1024, HH = 4, DH = 64;
constexpr int  NROW = BB * TT;                       // 16384 rows of (b,t)
constexpr int  HD   = HH * DH;                       // 256
constexpr long Y_ELEMS  = (long)NROW * DD;           // 16,777,216
constexpr long KV_ELEMS = (long)BB * HH * DH * DH;   // 65,536
constexpr float EPSV = 1e-6f;

// ---------------------------------------------------------------------------
// WMMA fragment helpers (wave32, v_wmma_f32_16x16x32_f16 layouts per CDNA5 ISA)
// ---------------------------------------------------------------------------
DEV int frag_k(int i, int hi) {
  int v = i >> 1, hb = i & 1;
  int g = v >> 2;
  int j = ((v & 3) << 1) | hb;
  return g * 16 + hi * 8 + j;
}

// A fragment (16x32, MxK) from a row-major f16 matrix, contiguous in k.
// Also serves as the B fragment when B = W^T with W stored row-major [N,K].
DEV v16h load_frag_rowmajor(const _Float16* p, int ld) {
  int lane = threadIdx.x & 31;
  int m = lane & 15, hi = lane >> 4;
  const _Float16* r = p + (long)m * ld + hi * 8;
  v8h a = *(const v8h*)(r);        // k = hi*8 + 0..7
  v8h b = *(const v8h*)(r + 16);   // k = 16 + hi*8 + 0..7
  v16h o;
#pragma unroll
  for (int i = 0; i < 8; ++i) { o[i] = a[i]; o[8 + i] = b[i]; }
  return o;
}

// Generic strided f16 fragment loader with k bound (pads with zero).
DEV v16h load_frag_strided_h(const _Float16* p, int mstride, int kstride, int kmax) {
  int lane = threadIdx.x & 31;
  int m = lane & 15, hi = lane >> 4;
  v16h o;
#pragma unroll
  for (int i = 0; i < 16; ++i) {
    int k = frag_k(i, hi);
    o[i] = (k < kmax) ? p[m * mstride + k * kstride] : (_Float16)0.0f;
  }
  return o;
}

// Same but converting from an f32 source (reads the f32 S state in LDS).
DEV v16h load_frag_strided_f32(const float* p, int mstride, int kstride) {
  int lane = threadIdx.x & 31;
  int m = lane & 15, hi = lane >> 4;
  v16h o;
#pragma unroll
  for (int i = 0; i < 16; ++i) {
    int k = frag_k(i, hi);
    o[i] = (_Float16)p[m * mstride + k * kstride];
  }
  return o;
}

DEV v8f load_cfrag(const float* p, int ld) {
  int lane = threadIdx.x & 31;
  int n = lane & 15, hi = lane >> 4;
  v8f c;
#pragma unroll
  for (int r = 0; r < 8; ++r) c[r] = p[(r + hi * 8) * ld + n];
  return c;
}

DEV void store_cfrag(float* p, int ld, v8f c) {
  int lane = threadIdx.x & 31;
  int n = lane & 15, hi = lane >> 4;
#pragma unroll
  for (int r = 0; r < 8; ++r) p[(r + hi * 8) * ld + n] = c[r];
}

DEV v8f wmma_f16(v16h a, v16h b, v8f c) {
  return __builtin_amdgcn_wmma_f32_16x16x32_f16(
      /*neg_a=*/false, a, /*neg_b=*/false, b,
      /*c_mod=*/(short)0, c, /*reuse_a=*/false, /*reuse_b=*/false);
}

// ---------------------------------------------------------------------------
// Async global->LDS staging (CDNA5 GLOBAL_LOAD_ASYNC_TO_LDS, ASYNCcnt-tracked)
// Copies a 16x64 f16 tile (row stride HD halves in global, 64 in LDS).
// ---------------------------------------------------------------------------
DEV void async_tile_to_lds(const _Float16* gsrc, _Float16* ldst, int lane) {
  unsigned l0 = (unsigned)(uintptr_t)ldst;   // low 32 bits == LDS offset
#pragma unroll
  for (int it = 0; it < 4; ++it) {
    int idx = it * 32 + lane;
    int t = idx >> 3, seg = idx & 7;         // row t, 16B segment within row
    const void* g = (const void*)((const char*)gsrc + (long)t * (HD * 2) + seg * 16);
    unsigned l = l0 + (unsigned)(t * 128 + seg * 16);
    asm volatile("global_load_async_to_lds_b128 %0, %1, off"
                 :: "v"(l), "v"(g) : "memory");
  }
}

DEV void wait_async_zero() {
  asm volatile("s_wait_asynccnt 0x0" ::: "memory");
}

// ---------------------------------------------------------------------------
// Kernels
// ---------------------------------------------------------------------------
__global__ void cast_f32_f16(const float* __restrict__ in, _Float16* __restrict__ out, long n) {
  long i = (long)blockIdx.x * blockDim.x + threadIdx.x;
  if (i < n) out[i] = (_Float16)in[i];
}

// C[M,N] = A[M,K] (f16 row-major) * W[N,K]^T (f16 row-major), f32 accumulate.
// Register-blocked: wave tile 32x64 (2x4 subtiles), block tile 64x256 (8 waves).
// Requires M % 64 == 0, N % 256 == 0, K % 32 == 0.
template <bool OUT_F32>
__global__ __launch_bounds__(256)
void gemm_wmma(const _Float16* __restrict__ A, const _Float16* __restrict__ W,
               void* __restrict__ Cout, int M, int N, int K) {
  int w = threadIdx.x >> 5;
  int wm = w >> 2;                 // 0..1
  int wn = w & 3;                  // 0..3
  int nBlocksN = N >> 8;           // N / 256
  int bm = blockIdx.x / nBlocksN;
  int bn = blockIdx.x % nBlocksN;
  long row0 = (long)bm * 64 + wm * 32;
  long col0 = (long)bn * 256 + wn * 64;

  const _Float16* a0 = A + row0 * K;
  const _Float16* b0 = W + col0 * K;

  v8f acc[2][4] = {};
  for (int k = 0; k < K; k += 32) {
    if (k + 64 < K) {                         // CDNA5 global_prefetch_b8
      __builtin_prefetch(a0 + k + 64, 0, 0);
      __builtin_prefetch(a0 + (long)16 * K + k + 64, 0, 0);
      __builtin_prefetch(b0 + k + 64, 0, 0);
      __builtin_prefetch(b0 + (long)32 * K + k + 64, 0, 0);
    }
    v16h af[2], bf[4];
#pragma unroll
    for (int i = 0; i < 2; ++i)
      af[i] = load_frag_rowmajor(a0 + (long)i * 16 * K + k, K);
#pragma unroll
    for (int j = 0; j < 4; ++j)
      bf[j] = load_frag_rowmajor(b0 + (long)j * 16 * K + k, K);
#pragma unroll
    for (int i = 0; i < 2; ++i)
#pragma unroll
      for (int j = 0; j < 4; ++j)
        acc[i][j] = wmma_f16(af[i], bf[j], acc[i][j]);
  }

  int lane = threadIdx.x & 31;
  int n = lane & 15, hi = lane >> 4;
#pragma unroll
  for (int i = 0; i < 2; ++i)
#pragma unroll
    for (int j = 0; j < 4; ++j) {
      long cbase = (row0 + i * 16) * N + col0 + j * 16 + n;
      if (OUT_F32) {
        float* C = (float*)Cout;
#pragma unroll
        for (int r = 0; r < 8; ++r) C[cbase + (long)(r + hi * 8) * N] = acc[i][j][r];
      } else {
        _Float16* C = (_Float16*)Cout;
#pragma unroll
        for (int r = 0; r < 8; ++r) C[cbase + (long)(r + hi * 8) * N] = (_Float16)acc[i][j][r];
      }
    }
}

// Feature map: p = elu(x)+1, normalized to unit L2 over d=64, in place.
__global__ void featmap_kernel(_Float16* __restrict__ q, _Float16* __restrict__ k) {
  long idx = (long)blockIdx.x * blockDim.x + threadIdx.x;
  if (idx >= (long)NROW * HH) return;
  long base = (idx >> 2) * HD + (idx & 3) * DH;
  _Float16* ptrs[2] = {q, k};
#pragma unroll
  for (int which = 0; which < 2; ++which) {
    _Float16* p = ptrs[which] + base;
    float tmp[DH];
    float ss = 0.f;
    for (int i = 0; i < DH; ++i) {
      float x = (float)p[i];
      float e = (x > 0.f) ? (x + 1.f) : __expf(x);
      tmp[i] = e;
      ss += e * e;
    }
    float inv = __builtin_amdgcn_rcpf(__builtin_sqrtf(ss) + EPSV);
    for (int i = 0; i < DH; ++i) p[i] = (_Float16)(tmp[i] * inv);
  }
}

// Chunked causal linear-attention scan. One wave per (b,h). Chunk = 16 steps.
__global__ void __launch_bounds__(32)
scan_kernel(const _Float16* __restrict__ q16, const _Float16* __restrict__ k16,
            const _Float16* __restrict__ v16, _Float16* __restrict__ y16,
            float* __restrict__ kv_out, float* __restrict__ kc_out) {
  __shared__ float    S[DH * DH];      // 16 KB: S[d][j] = KV^T
  __shared__ float    kcum[DH];
  __shared__ _Float16 qs[16 * DH];     // chunk tiles [t][d]
  __shared__ _Float16 ks[16 * DH];
  __shared__ _Float16 vs[16 * DH];
  __shared__ _Float16 aq[16 * 32];     // masked Aqk, padded to K=32
  __shared__ float    den[16];
  __shared__ float    red[32];

  const int lane = threadIdx.x;
  const int bh = blockIdx.x;
  const int b = bh >> 2, h = bh & 3;
  const long rowbase = (long)b * TT;
  const int hoff = h * DH;

  for (int i = lane; i < DH * DH; i += 32) S[i] = 0.f;
  for (int i = lane; i < DH; i += 32) kcum[i] = 0.f;
  for (int i = lane; i < 16 * 16; i += 32) {     // zero pad cols 16..31 once
    int m = i >> 4, n2 = i & 15;
    aq[m * 32 + 16 + n2] = (_Float16)0.0f;
  }
  __syncthreads();

  for (int c = 0; c < TT / 16; ++c) {
    const int t0 = c * 16;
    const long gchunk = (rowbase + t0) * (long)HD + hoff;
    // Async stage chunk into LDS (ASYNCcnt), then wait.
    async_tile_to_lds(q16 + gchunk, qs, lane);
    async_tile_to_lds(k16 + gchunk, ks, lane);
    async_tile_to_lds(v16 + gchunk, vs, lane);
    wait_async_zero();
    __syncthreads();

    // Aqk[t][s] = q_t . k_s  (two 16x16x32 wmma over d=64)
    v8f aqk = {};
    {
      v16h a  = load_frag_rowmajor(qs, DH);
      v16h bb = load_frag_rowmajor(ks, DH);      // B[k=d][n=s] = K[s][d]
      aqk = wmma_f16(a, bb, aqk);
      a  = load_frag_rowmajor(qs + 32, DH);
      bb = load_frag_rowmajor(ks + 32, DH);
      aqk = wmma_f16(a, bb, aqk);
    }

    // den[t] = q_t . kcum_prev
    {
      int m = lane & 15, hi = lane >> 4;
      float s = 0.f;
      for (int d = hi * 32; d < hi * 32 + 32; ++d)
        s += (float)qs[m * DH + d] * kcum[d];
      red[lane] = s;
      __syncthreads();
      if (lane < 16) den[lane] = red[lane] + red[lane + 16];
      __syncthreads();
    }

    // Mask Aqk causally (s<=t), stash as f16 A-operand, row-sum -> den
    {
      int n = lane & 15, hi = lane >> 4;
#pragma unroll
      for (int r = 0; r < 8; ++r) {
        int m = r + hi * 8;
        float val = (n <= m) ? aqk[r] : 0.f;
        aq[m * 32 + n] = (_Float16)val;
        atomicAdd(&den[m], val);                 // ds_add_f32
      }
    }
    __syncthreads();

    // Reciprocal denominator (v_rcp_f32) + kcum += sum_t k[t][.]
    if (lane < 16) den[lane] = __builtin_amdgcn_rcpf(den[lane] + EPSV);
    for (int d = lane; d < DH; d += 32) {
      float s = kcum[d];
      for (int t = 0; t < 16; ++t) s += (float)ks[t * DH + d];
      kcum[d] = s;
    }
    __syncthreads();

    // Y = Q * S_prev + mask(Aqk) * V, scaled by 1/den
    v16h aA0 = load_frag_rowmajor(qs, DH);
    v16h aA1 = load_frag_rowmajor(qs + 32, DH);
    v16h aM  = load_frag_rowmajor(aq, 32);
    for (int jt = 0; jt < 4; ++jt) {
      v8f y = {};
      v16h bS0 = load_frag_strided_f32(S + jt * 16, 1, DH);
      y = wmma_f16(aA0, bS0, y);
      v16h bS1 = load_frag_strided_f32(S + 32 * DH + jt * 16, 1, DH);
      y = wmma_f16(aA1, bS1, y);
      v16h bV  = load_frag_strided_h(vs + jt * 16, 1, DH, 16);
      y = wmma_f16(aM, bV, y);

      int n = lane & 15, hi = lane >> 4;
#pragma unroll
      for (int r = 0; r < 8; ++r) {
        int m = r + hi * 8;
        long g = (rowbase + t0 + m) * (long)HD + hoff + jt * 16 + n;
        y16[g] = (_Float16)(y[r] * den[m]);
      }
    }
    __syncthreads();

    // S += K^T V
    for (int di = 0; di < 4; ++di) {
      v16h aK = load_frag_strided_h(ks + di * 16, 1, DH, 16);
      for (int ji = 0; ji < 4; ++ji) {
        v16h bV = load_frag_strided_h(vs + ji * 16, 1, DH, 16);
        v8f cc = load_cfrag(S + (di * 16) * DH + ji * 16, DH);
        cc = wmma_f16(aK, bV, cc);
        store_cfrag(S + (di * 16) * DH + ji * 16, DH, cc);
      }
    }
    __syncthreads();
  }

  // Final states: kv_f[b,h,j,d] = S[d][j] ; kc_f[b,h,d] = kcum[d]
  float* kv = kv_out + (long)bh * DH * DH;
  for (int i = lane; i < DH * DH; i += 32) {
    int j = i >> 6, dd = i & 63;
    kv[i] = S[dd * DH + j];
  }
  float* kc = kc_out + (long)bh * DH;
  for (int i = lane; i < DH; i += 32) kc[i] = kcum[i];
}

// ---------------------------------------------------------------------------
// Host-side launcher
// ---------------------------------------------------------------------------
extern "C" void kernel_launch(void* const* d_in, const int* in_sizes, int n_in,
                              void* d_out, int out_size, void* d_ws, size_t ws_size,
                              hipStream_t stream) {
  const float* x  = (const float*)d_in[0];
  const float* Wq = (const float*)d_in[1];
  const float* Wk = (const float*)d_in[2];
  const float* Wv = (const float*)d_in[3];
  const float* Wo = (const float*)d_in[4];

  _Float16* x16  = (_Float16*)d_ws;
  _Float16* wq16 = x16  + (long)NROW * DD;
  _Float16* wk16 = wq16 + (long)HD * DD;
  _Float16* wv16 = wk16 + (long)HD * DD;
  _Float16* wo16 = wv16 + (long)HD * DD;
  _Float16* q16  = wo16 + (long)DD * HD;
  _Float16* k16  = q16  + (long)NROW * HD;
  _Float16* v16  = k16  + (long)NROW * HD;
  _Float16* y16  = v16  + (long)NROW * HD;

  // 1) fp32 -> fp16 casts
  {
    long n = (long)NROW * DD;
    cast_f32_f16<<<(int)((n + 255) / 256), 256, 0, stream>>>(x, x16, n);
    long nw = (long)HD * DD;
    int gw = (int)((nw + 255) / 256);
    cast_f32_f16<<<gw, 256, 0, stream>>>(Wq, wq16, nw);
    cast_f32_f16<<<gw, 256, 0, stream>>>(Wk, wk16, nw);
    cast_f32_f16<<<gw, 256, 0, stream>>>(Wv, wv16, nw);
    cast_f32_f16<<<gw, 256, 0, stream>>>(Wo, wo16, nw);
  }

  // 2) QKV projections: [16384,1024] x [1024,256]^T -> f16
  {
    int blocks = (NROW / 64) * (HD / 256);   // 256
    gemm_wmma<false><<<blocks, 256, 0, stream>>>(x16, wq16, q16, NROW, HD, DD);
    gemm_wmma<false><<<blocks, 256, 0, stream>>>(x16, wk16, k16, NROW, HD, DD);
    gemm_wmma<false><<<blocks, 256, 0, stream>>>(x16, wv16, v16, NROW, HD, DD);
  }

  // 3) Feature map in place on q16/k16
  {
    long n = (long)NROW * HH;
    featmap_kernel<<<(int)((n + 255) / 256), 256, 0, stream>>>(q16, k16);
  }

  // 4) Chunked causal scan -> y16, final states into d_out tail
  {
    float* out = (float*)d_out;
    scan_kernel<<<BB * HH, 32, 0, stream>>>(q16, k16, v16, y16,
                                            out + Y_ELEMS,
                                            out + Y_ELEMS + KV_ELEMS);
  }

  // 5) Output projection: [16384,256] x [256,1024]^T -> f32 into d_out
  {
    int blocks = (NROW / 64) * (DD / 256);   // 1024
    gemm_wmma<true><<<blocks, 256, 0, stream>>>(y16, wo16, d_out, NROW, DD, HD);
  }
}
